// ImprovedDecoder_85847806312497
// MI455X (gfx1250) — compile-verified, matrix operand order checked by reading, not run on hardware
//
#include <hip/hip_runtime.h>
#include <hip/hip_bf16.h>
#include <math.h>

// ---------------------------------------------------------------------------
// CDNA5 (gfx1250) WMMA types & helpers
// ---------------------------------------------------------------------------
typedef __attribute__((ext_vector_type(16))) __bf16    v16bf;
typedef __attribute__((ext_vector_type(2)))  __bf16    v2bf;
typedef __attribute__((ext_vector_type(8)))  float     v8f;
typedef __attribute__((ext_vector_type(4)))  unsigned  u32x4;

#if defined(__AMDGCN__) && __has_builtin(__builtin_amdgcn_cvt_pk_bf16_f32)
// single-instruction packed f32 -> 2xbf16
__device__ __forceinline__ unsigned pack2(float a, float b) {
    return __builtin_bit_cast(unsigned, __builtin_amdgcn_cvt_pk_bf16_f32(a, b));
}
#elif defined(__AMDGCN__) && __has_builtin(__builtin_amdgcn_perm)
// round-half-up to bf16 then assemble both high halves with one v_perm_b32
__device__ __forceinline__ unsigned pack2(float a, float b) {
    unsigned ua = __float_as_uint(a) + 0x8000u;
    unsigned ub = __float_as_uint(b) + 0x8000u;
    // result byte0,1 = ua bytes 2,3 ; byte2,3 = ub bytes 2,3
    return __builtin_amdgcn_perm(ub, ua, 0x07060302u);
}
#else
__device__ __forceinline__ unsigned short f32_to_bf16_rne(float f) {
    unsigned u = __float_as_uint(f);
    u += 0x7FFFu + ((u >> 16) & 1u);
    return (unsigned short)(u >> 16);
}
__device__ __forceinline__ unsigned pack2(float a, float b) {
    return (unsigned)f32_to_bf16_rne(a) | ((unsigned)f32_to_bf16_rne(b) << 16);
}
#endif

__device__ __forceinline__ float sigmoidf_(float x) {
    return 1.0f / (1.0f + __expf(-x));
}

// Fragment load from LDS: row of 32 bf16 stored as 16 u32.
// hi=0 needs K {0..7,16..23} -> u32 {0..3, 8..11}; hi=1 -> {4..7,12..15}
struct bfrag_t { u32x4 a, b; };
__device__ __forceinline__ v16bf load_frag(const unsigned* base, int hi) {
    bfrag_t f;
    f.a = *(const u32x4*)(base + hi * 4);
    f.b = *(const u32x4*)(base + 8 + hi * 4);
    return __builtin_bit_cast(v16bf, f);
}

// Model dimensions
#define DB   64      // batch
#define DS   256     // src len
#define DV   32000   // vocab
#define DE   512     // embed
#define DH   512     // hidden
#define DA   512     // attention dim
#define D2H  1024    // 2*H

// ---------------------------------------------------------------------------
// Generic WMMA GEMM:  C[64,N] (+)= A[64,K] * B[N,K]^T + bias[N]
//   block = 256 thr (8 waves); tile = 64(M) x 128(N); K streamed by 32,
//   LDS double-buffered (one barrier per K-step).
//   grid = (N/128). Requires K%32==0, N%128==0, M==64, 16B-aligned A,B.
// ---------------------------------------------------------------------------
__global__ __launch_bounds__(256) void k_wmma_gemm(
    const float* __restrict__ A, int lda,
    const float* __restrict__ B, int ldb,
    const float* __restrict__ bias,
    float* __restrict__ C, int ldc,
    int K, int accumulate)
{
    __shared__ unsigned As_u[2][64 * 16];    // 2 x 4 KB
    __shared__ unsigned Bs_u[2][128 * 16];   // 2 x 8 KB

    const int tid  = threadIdx.x;
    const int wave = tid >> 5;
    const int lane = tid & 31;
    const int hi   = lane >> 4;
    const int ml   = lane & 15;
    const int n0   = blockIdx.x * 128;

    auto stage = [&](int buf, int k0) {
        {   // A: thread t -> row t>>2, cols (t&3)*8 .. +8  (2x float4)
            int r = tid >> 2, q = tid & 3;
            const float4* pa = (const float4*)(A + (size_t)r * lda + k0 + q * 8);
            float4 a0 = pa[0], a1 = pa[1];
            u32x4 pk;
            pk[0] = pack2(a0.x, a0.y); pk[1] = pack2(a0.z, a0.w);
            pk[2] = pack2(a1.x, a1.y); pk[3] = pack2(a1.z, a1.w);
            *(u32x4*)&As_u[buf][r * 16 + q * 4] = pk;
        }
        {   // B: thread t -> row t&127, cols (t>>7)*16 .. +16 (4x float4)
            int r = tid & 127, cg = tid >> 7;
            const float4* pb = (const float4*)(B + (size_t)(n0 + r) * ldb + k0 + cg * 16);
            float4 b0 = pb[0], b1 = pb[1], b2 = pb[2], b3 = pb[3];
            u32x4 p0, p1;
            p0[0] = pack2(b0.x, b0.y); p0[1] = pack2(b0.z, b0.w);
            p0[2] = pack2(b1.x, b1.y); p0[3] = pack2(b1.z, b1.w);
            p1[0] = pack2(b2.x, b2.y); p1[1] = pack2(b2.z, b2.w);
            p1[2] = pack2(b3.x, b3.y); p1[3] = pack2(b3.z, b3.w);
            *(u32x4*)&Bs_u[buf][r * 16 + cg * 8]     = p0;
            *(u32x4*)&Bs_u[buf][r * 16 + cg * 8 + 4] = p1;
        }
    };

    v8f acc[4];
    #pragma unroll
    for (int mt = 0; mt < 4; ++mt)
        #pragma unroll
        for (int e = 0; e < 8; ++e) acc[mt][e] = 0.0f;

    stage(0, 0);
    int cur = 0;
    for (int k0 = 0; k0 < K; k0 += 32) {
        __syncthreads();                       // buf[cur] ready; buf[cur^1] free
        if (k0 + 32 < K) stage(cur ^ 1, k0 + 32);
        v16bf bf_ = load_frag(&Bs_u[cur][(wave * 16 + ml) * 16], hi);
        #pragma unroll
        for (int mt = 0; mt < 4; ++mt) {
            v16bf af = load_frag(&As_u[cur][(mt * 16 + ml) * 16], hi);
            acc[mt] = __builtin_amdgcn_wmma_f32_16x16x32_bf16(
                          false, af, false, bf_, (short)0, acc[mt], false, false);
        }
        cur ^= 1;
    }

    // epilogue: D layout => n = lane&15, m = r + 8*hi (+16*mt)
    const int ng = n0 + wave * 16 + ml;
    const float bv = bias ? bias[ng] : 0.0f;
    #pragma unroll
    for (int mt = 0; mt < 4; ++mt) {
        #pragma unroll
        for (int r = 0; r < 8; ++r) {
            int mg = mt * 16 + r + hi * 8;
            size_t off = (size_t)mg * ldc + ng;
            float v = acc[mt][r] + bv;
            if (accumulate) v += C[off];
            C[off] = v;
        }
    }
}

// ---------------------------------------------------------------------------
// Fused attention-score kernel.
//  block = (s-tile of 16, batch b):
//    comb[16,512] = enc[b, s0:s0+16, :] @ Wenc^T + b_enc + dec_proj[b]
//  via double-buffered WMMA (8 waves x 4 n-subtiles), then per-row LN +
//  tanh + dot(v_att) done entirely with wave shuffles (no barriers).
// ---------------------------------------------------------------------------
__global__ __launch_bounds__(256) void k_attn_scores(
    const float* __restrict__ enc,       // [B,S,2H]
    const float* __restrict__ Wenc,      // [A,2H]
    const float* __restrict__ b_enc,     // [A]
    const float* __restrict__ dec_proj,  // [B,A]
    const float* __restrict__ ln_g,
    const float* __restrict__ ln_b,
    const float* __restrict__ v_att,     // [A]
    const int*   __restrict__ mask,      // [B,S]
    float* __restrict__ scores)          // [B,S]
{
    __shared__ unsigned As_u[2][16 * 16];              // 2 x 1 KB
    __shared__ union {
        unsigned bs[2][512 * 16];   // 2 x 32 KB
        float comb[16 * 512];       // 32 KB (reused after GEMM)
    } sh;

    const int tid  = threadIdx.x;
    const int wave = tid >> 5;
    const int lane = tid & 31;
    const int hi   = lane >> 4;
    const int ml   = lane & 15;
    const int b    = blockIdx.y;
    const int s0   = blockIdx.x * 16;
    const float* Arow = enc + ((size_t)b * DS + s0) * D2H;

    auto stage = [&](int buf, int k0) {
        {   // A: thread t -> row t>>4, cols (t&15)*2 (float2)
            int r = tid >> 4, j = tid & 15;
            const float2* pa = (const float2*)(Arow + (size_t)r * D2H + k0 + j * 2);
            float2 v = *pa;
            As_u[buf][r * 16 + j] = pack2(v.x, v.y);
        }
        #pragma unroll
        for (int rr = 0; rr < 2; ++rr) {   // B: thread t -> rows 2t,2t+1, 32 cols
            int r = tid * 2 + rr;
            const float4* pb = (const float4*)(Wenc + (size_t)r * D2H + k0);
            #pragma unroll
            for (int q = 0; q < 2; ++q) {
                float4 b0 = pb[q * 4 + 0], b1 = pb[q * 4 + 1];
                float4 b2 = pb[q * 4 + 2], b3 = pb[q * 4 + 3];
                u32x4 p0, p1;
                p0[0] = pack2(b0.x, b0.y); p0[1] = pack2(b0.z, b0.w);
                p0[2] = pack2(b1.x, b1.y); p0[3] = pack2(b1.z, b1.w);
                p1[0] = pack2(b2.x, b2.y); p1[1] = pack2(b2.z, b2.w);
                p1[2] = pack2(b3.x, b3.y); p1[3] = pack2(b3.z, b3.w);
                *(u32x4*)&sh.bs[buf][r * 16 + q * 8]     = p0;
                *(u32x4*)&sh.bs[buf][r * 16 + q * 8 + 4] = p1;
            }
        }
    };

    v8f acc[4];
    #pragma unroll
    for (int i = 0; i < 4; ++i)
        #pragma unroll
        for (int e = 0; e < 8; ++e) acc[i][e] = 0.0f;

    stage(0, 0);
    int cur = 0;
    for (int k0 = 0; k0 < D2H; k0 += 32) {
        __syncthreads();
        if (k0 + 32 < D2H) stage(cur ^ 1, k0 + 32);
        v16bf af = load_frag(&As_u[cur][ml * 16], hi);
        #pragma unroll
        for (int sub = 0; sub < 4; ++sub) {
            int n = wave * 64 + sub * 16 + ml;
            v16bf bfr = load_frag(&sh.bs[cur][n * 16], hi);
            acc[sub] = __builtin_amdgcn_wmma_f32_16x16x32_bf16(
                           false, af, false, bfr, (short)0, acc[sub], false, false);
        }
        cur ^= 1;
    }
    __syncthreads();   // all waves done reading bs before comb overlays it

    // dump accumulators (+ dec_proj + b_enc) to LDS as comb[16][512]
    #pragma unroll
    for (int sub = 0; sub < 4; ++sub) {
        int n = wave * 64 + sub * 16 + ml;
        float add = dec_proj[b * DA + n] + b_enc[n];
        #pragma unroll
        for (int r = 0; r < 8; ++r)
            sh.comb[(r + hi * 8) * DA + n] = acc[sub][r] + add;
    }
    __syncthreads();

    // per-row LN + tanh + dot(v_att): 16 rows x 16 lanes, shuffle reductions
    {
        const int r = tid >> 4;      // row 0..15
        const int j = tid & 15;      // lane-in-row
        float xv[32];
        float s1 = 0.0f, s2 = 0.0f;
        #pragma unroll
        for (int k = 0; k < 32; ++k) {
            float x = sh.comb[r * DA + j + k * 16];
            xv[k] = x; s1 += x; s2 += x * x;
        }
        #pragma unroll
        for (int m = 1; m < 16; m <<= 1) {
            s1 += __shfl_xor(s1, m, 32);
            s2 += __shfl_xor(s2, m, 32);
        }
        float mean = s1 * (1.0f / DA);
        float inv  = rsqrtf(s2 * (1.0f / DA) - mean * mean + 1e-5f);
        float dot = 0.0f;
        #pragma unroll
        for (int k = 0; k < 32; ++k) {
            int n = j + k * 16;
            float t = tanhf((xv[k] - mean) * inv * ln_g[n] + ln_b[n]);
            dot += t * v_att[n];
        }
        #pragma unroll
        for (int m = 1; m < 16; m <<= 1) dot += __shfl_xor(dot, m, 32);
        if (j == 0) {
            int s = s0 + r;
            float sc = dot;
            if (mask[b * DS + s] == 0) sc = -__builtin_inff();
            scores[b * DS + s] = sc;
        }
    }
}

// ---------------------------------------------------------------------------
// Embedding gather + LayerNorm -> xemb[B,E]
// ---------------------------------------------------------------------------
__global__ __launch_bounds__(256) void k_embed_ln(
    const int* __restrict__ tok, const float* __restrict__ emb,
    const float* __restrict__ g, const float* __restrict__ bb,
    float* __restrict__ out)
{
    __shared__ float red[256], red2[256];
    int tid = threadIdx.x, b = blockIdx.x;
    const float* row = emb + (size_t)tok[b] * DE;
    float x1 = row[tid], x2 = row[tid + 256];
    red[tid] = x1 + x2; red2[tid] = x1 * x1 + x2 * x2;
    __syncthreads();
    for (int st = 128; st > 0; st >>= 1) {
        if (tid < st) { red[tid] += red[tid + st]; red2[tid] += red2[tid + st]; }
        __syncthreads();
    }
    float mean = red[0] * (1.0f / DE);
    float var  = red2[0] * (1.0f / DE) - mean * mean;
    float inv  = rsqrtf(var + 1e-5f);
    out[b * DE + tid]       = (x1 - mean) * inv * g[tid]       + bb[tid];
    out[b * DE + tid + 256] = (x2 - mean) * inv * g[tid + 256] + bb[tid + 256];
}

// ---------------------------------------------------------------------------
// Softmax over S + context = attn_w @ enc   (one block per batch element)
// ---------------------------------------------------------------------------
__global__ __launch_bounds__(256) void k_softmax_context(
    const float* __restrict__ scores, const float* __restrict__ enc,
    float* __restrict__ attn_w, float* __restrict__ context)
{
    __shared__ float red[256];
    __shared__ float wbuf[256];
    int tid = threadIdx.x, b = blockIdx.x;
    float sc = scores[b * DS + tid];
    red[tid] = sc; __syncthreads();
    for (int st = 128; st > 0; st >>= 1) {
        if (tid < st) red[tid] = fmaxf(red[tid], red[tid + st]);
        __syncthreads();
    }
    float mx = red[0]; __syncthreads();
    float e = __expf(sc - mx);
    red[tid] = e; __syncthreads();
    for (int st = 128; st > 0; st >>= 1) {
        if (tid < st) red[tid] += red[tid + st];
        __syncthreads();
    }
    float w = e / red[0];
    attn_w[b * DS + tid] = w;
    wbuf[tid] = w;
    __syncthreads();

    const float* eb = enc + (size_t)b * DS * D2H;
    for (int k = tid; k < D2H; k += 256) {
        float a = 0.0f;
        for (int s = 0; s < DS; ++s) a += wbuf[s] * eb[(size_t)s * D2H + k];
        context[b * D2H + k] = a;
    }
}

// ---------------------------------------------------------------------------
// LSTM cell pointwise: gates[B,4H] (i,f,g,o), c_prev[B,H] -> h,c
// ---------------------------------------------------------------------------
__global__ __launch_bounds__(256) void k_lstm_cell(
    const float* __restrict__ gates, const float* __restrict__ c_prev,
    float* __restrict__ h_out, float* __restrict__ c_out)
{
    int idx = blockIdx.x * 256 + threadIdx.x;   // B*H = 32768
    int b = idx >> 9, j = idx & 511;
    size_t base = (size_t)b * (4 * DH);
    float i = sigmoidf_(gates[base + j]);
    float f = sigmoidf_(gates[base + DH + j]);
    float g = tanhf    (gates[base + 2 * DH + j]);
    float o = sigmoidf_(gates[base + 3 * DH + j]);
    float c = f * c_prev[idx] + i * g;
    c_out[idx] = c;
    h_out[idx] = o * tanhf(c);
}

// ---------------------------------------------------------------------------
// final = LN(h_l1)*g+b + proj_ctx      (one block per batch element)
// ---------------------------------------------------------------------------
__global__ __launch_bounds__(256) void k_ln_add(
    const float* __restrict__ h, const float* __restrict__ g,
    const float* __restrict__ bb, const float* __restrict__ proj,
    float* __restrict__ out)
{
    __shared__ float red[256], red2[256];
    int tid = threadIdx.x, b = blockIdx.x;
    float x1 = h[b * DH + tid], x2 = h[b * DH + tid + 256];
    red[tid] = x1 + x2; red2[tid] = x1 * x1 + x2 * x2;
    __syncthreads();
    for (int st = 128; st > 0; st >>= 1) {
        if (tid < st) { red[tid] += red[tid + st]; red2[tid] += red2[tid + st]; }
        __syncthreads();
    }
    float mean = red[0] * (1.0f / DH);
    float var  = red2[0] * (1.0f / DH) - mean * mean;
    float inv  = rsqrtf(var + 1e-5f);
    out[b * DH + tid]       = (x1 - mean) * inv * g[tid]       + bb[tid]       + proj[b * DH + tid];
    out[b * DH + tid + 256] = (x2 - mean) * inv * g[tid + 256] + bb[tid + 256] + proj[b * DH + tid + 256];
}

// ---------------------------------------------------------------------------
// Host-side orchestration
// ---------------------------------------------------------------------------
extern "C" void kernel_launch(void* const* d_in, const int* in_sizes, int n_in,
                              void* d_out, int out_size, void* d_ws, size_t ws_size,
                              hipStream_t stream)
{
    const int*   tok      = (const int*)  d_in[0];
    const float* h0       = (const float*)d_in[1];   // [2,B,H]
    const float* c0       = (const float*)d_in[2];   // [2,B,H]
    const float* enc      = (const float*)d_in[3];   // [B,S,2H]
    const int*   mask     = (const int*)  d_in[4];   // [B,S]
    const float* emb      = (const float*)d_in[5];
    const float* ln_emb_g = (const float*)d_in[6];
    const float* ln_emb_b = (const float*)d_in[7];
    const float* W_enc    = (const float*)d_in[8];   // [A,2H]
    const float* b_enc    = (const float*)d_in[9];
    const float* W_dec    = (const float*)d_in[10];  // [A,H]
    const float* b_dec    = (const float*)d_in[11];
    const float* v_att    = (const float*)d_in[12];  // [1,A]
    const float* ln_att_g = (const float*)d_in[13];
    const float* ln_att_b = (const float*)d_in[14];
    const float* W_ih0    = (const float*)d_in[15];  // [4H, E+2H]
    const float* W_hh0    = (const float*)d_in[16];  // [4H, H]
    const float* b_ih0    = (const float*)d_in[17];
    const float* b_hh0    = (const float*)d_in[18];
    const float* W_ih1    = (const float*)d_in[19];  // [4H, H]
    const float* W_hh1    = (const float*)d_in[20];
    const float* b_ih1    = (const float*)d_in[21];
    const float* b_hh1    = (const float*)d_in[22];
    const float* ln_l_g   = (const float*)d_in[23];
    const float* ln_l_b   = (const float*)d_in[24];
    const float* W_ctx    = (const float*)d_in[25];  // [H,2H]
    const float* b_ctx    = (const float*)d_in[26];
    const float* W_out    = (const float*)d_in[27];  // [V,H]
    const float* b_out    = (const float*)d_in[28];

    // outputs: logits[B,1,V], h_new[2,B,H], c_new[2,B,H], attn_w[B,S]
    float* out    = (float*)d_out;
    float* logits = out;
    float* h_new  = logits + (size_t)DB * DV;
    float* c_new  = h_new + 2 * DB * DH;
    float* attn_w = c_new + 2 * DB * DH;
    float* h_l0 = h_new;              float* h_l1 = h_new + DB * DH;
    float* c_l0 = c_new;              float* c_l1 = c_new + DB * DH;

    // workspace layout (floats)
    float* w = (float*)d_ws;
    float* xemb     = w;                  // 64*512
    float* dec_proj = xemb + DB * DE;     // 64*512
    float* scoresb  = dec_proj + DB * DA; // 64*256
    float* context  = scoresb + DB * DS;  // 64*1024
    float* gates0   = context + DB * D2H; // 64*2048
    float* gates1   = gates0 + DB * 4 * DH;
    float* proj_ctx = gates1 + DB * 4 * DH;
    float* final_in = proj_ctx + DB * DH;
    (void)in_sizes; (void)n_in; (void)out_size; (void)ws_size;

    const dim3 blk(256);

    // 1) embedding + LN
    k_embed_ln<<<DB, blk, 0, stream>>>(tok, emb, ln_emb_g, ln_emb_b, xemb);

    // 2) dec_proj = h0[1] @ W_dec^T + b_dec     (N=512, K=512)
    k_wmma_gemm<<<dim3(DA / 128), blk, 0, stream>>>(
        h0 + DB * DH, DH, W_dec, DH, b_dec, dec_proj, DA, DH, 0);

    // 3) fused enc_proj + LN + tanh + v_att dot + mask -> scores
    k_attn_scores<<<dim3(DS / 16, DB), blk, 0, stream>>>(
        enc, W_enc, b_enc, dec_proj, ln_att_g, ln_att_b, v_att, mask, scoresb);

    // 4) softmax + context (also emits attn_w output)
    k_softmax_context<<<DB, blk, 0, stream>>>(scoresb, enc, attn_w, context);

    // 5) layer-0 gates = [xemb|context] @ W_ih0^T + h0[0] @ W_hh0^T + biases
    k_wmma_gemm<<<dim3(4 * DH / 128), blk, 0, stream>>>(
        xemb, DE, W_ih0, DE + D2H, b_ih0, gates0, 4 * DH, DE, 0);
    k_wmma_gemm<<<dim3(4 * DH / 128), blk, 0, stream>>>(
        context, D2H, W_ih0 + DE, DE + D2H, nullptr, gates0, 4 * DH, D2H, 1);
    k_wmma_gemm<<<dim3(4 * DH / 128), blk, 0, stream>>>(
        h0, DH, W_hh0, DH, b_hh0, gates0, 4 * DH, DH, 1);

    // 6) layer-0 cell
    k_lstm_cell<<<DB * DH / 256, blk, 0, stream>>>(gates0, c0, h_l0, c_l0);

    // 7) layer-1 gates = h_l0 @ W_ih1^T + h0[1] @ W_hh1^T + biases
    k_wmma_gemm<<<dim3(4 * DH / 128), blk, 0, stream>>>(
        h_l0, DH, W_ih1, DH, b_ih1, gates1, 4 * DH, DH, 0);
    k_wmma_gemm<<<dim3(4 * DH / 128), blk, 0, stream>>>(
        h0 + DB * DH, DH, W_hh1, DH, b_hh1, gates1, 4 * DH, DH, 1);

    // 8) layer-1 cell
    k_lstm_cell<<<DB * DH / 256, blk, 0, stream>>>(gates1, c0 + DB * DH, h_l1, c_l1);

    // 9) proj_ctx = context @ W_ctx^T + b_ctx   (N=512, K=1024)
    k_wmma_gemm<<<dim3(DH / 128), blk, 0, stream>>>(
        context, D2H, W_ctx, D2H, b_ctx, proj_ctx, DH, D2H, 0);

    // 10) final_in = LN(h_l1) + proj_ctx
    k_ln_add<<<DB, blk, 0, stream>>>(h_l1, ln_l_g, ln_l_b, proj_ctx, final_in);

    // 11) logits = final_in @ W_out^T + b_out   (N=32000, K=512)
    k_wmma_gemm<<<dim3(DV / 128), blk, 0, stream>>>(
        final_in, DH, W_out, DH, b_out, logits, DV, DH, 0);
}
